// NVFP4Linear_68951404970056
// MI455X (gfx1250) — compile-verified, matrix operand order checked by reading, not run on hardware
//
#include <hip/hip_runtime.h>

// NVFP4 fake-quant linear: y = dq(x) @ dq(w)^T  (fp32 out)
// Phase 1: quantize-dequantize both operands to f16 in d_ws (L2-resident, 42MB)
// Phase 2: WMMA f16 GEMM, K-step 32 == NVFP4 block size.
//          Wave tile 64x64 (16 WMMA per 16 b128 loads -> 1.0 loads/WMMA).

typedef __attribute__((ext_vector_type(16))) _Float16     v16h;
typedef __attribute__((ext_vector_type(8)))  float        v8f;
typedef __attribute__((ext_vector_type(4)))  float        v4f;
typedef __attribute__((ext_vector_type(4)))  unsigned int v4u;

#define GM 1024
#define GK 4096
#define GN 4096

// ------------------------------------------------------------------
// Phase 1: per-32-block NVFP4 quantize + dequantize  fp32 -> f16
// One thread per 32-element block (128B contiguous read, 64B write).
// ------------------------------------------------------------------
__global__ __launch_bounds__(256) void nvfp4_quantdq_kernel(
    const float* __restrict__ in, _Float16* __restrict__ out, int nblk)
{
  int b = blockIdx.x * 256 + threadIdx.x;
  if (b >= nblk) return;

  const v4f* p = (const v4f*)(in + (size_t)b * 32);
  float vals[32];
  #pragma unroll
  for (int g = 0; g < 8; ++g) {
    v4f t = p[g];
    #pragma unroll
    for (int i = 0; i < 4; ++i) vals[g * 4 + i] = t[i];
  }

  float bm = 1e-12f;
  #pragma unroll
  for (int i = 0; i < 32; ++i) bm = fmaxf(bm, fabsf(vals[i]));

  const float inv   = 6.0f / bm;          // |v|*inv = value on e2m1 grid scale
  const float scale = bm * (1.0f / 6.0f); // blockmax / NVFP4_MAX

  union H8 { _Float16 h[8]; v4u u; };
  v4u* dst = (v4u*)(out + (size_t)b * 32);
  #pragma unroll
  for (int g = 0; g < 4; ++g) {
    H8 o;
    #pragma unroll
    for (int i = 0; i < 8; ++i) {
      float v = vals[g * 8 + i];
      float s = fabsf(v) * inv;
      // nearest e2m1 palette value; ties round to the LOWER value
      // (jnp.argmin picks the first/smaller palette entry on a tie)
      float q = (s <= 0.25f) ? 0.0f
              : (s <= 0.75f) ? 0.5f
              : (s <= 1.25f) ? 1.0f
              : (s <= 1.75f) ? 1.5f
              : (s <= 2.50f) ? 2.0f
              : (s <= 3.50f) ? 3.0f
              : (s <= 5.00f) ? 4.0f
              :                6.0f;
      o.h[i] = (_Float16)copysignf(q * scale, v);
    }
    dst[g] = o.u;
  }
}

// ------------------------------------------------------------------
// Phase 2: f16 WMMA GEMM  C[GM,GN] = A[GM,GK] * B[GN,GK]^T
// Block: 128 threads = 4 waves (2x2), block tile 128x128.
// Wave tile: 64x64 = 4x4 WMMA 16x16 tiles, K-step 32.
// Per K-step: 16 global_load_b128 feed 16 v_wmma  (reuse = 2.0).
// ------------------------------------------------------------------
union Frag { v4u u[2]; v16h h; };

__global__ __launch_bounds__(128) void nvfp4_gemm_kernel(
    const _Float16* __restrict__ A,   // x_dq  [GM, GK] row-major
    const _Float16* __restrict__ B,   // w_dq  [GN, GK] row-major (row n = B-col n)
    float* __restrict__ C)            // [GM, GN]
{
  const int lane = threadIdx.x & 31;
  const int wave = threadIdx.x >> 5;
  const int wm   = wave >> 1;                       // 0..1
  const int wn   = wave & 1;                        // 0..1
  const int m_base = blockIdx.x * 128 + wm * 64;
  const int n_base = blockIdx.y * 128 + wn * 64;

  const int lr   = lane & 15;                       // row within 16-tile
  const int hi   = lane >> 4;                       // lane group 0/1
  const int koff = hi * 8;                          // K chunks: koff..+7, koff+16..+23

  // Per-lane row pointers (ISA 16-bit A-matrix 16x32 layout; B mirrors by N)
  const _Float16* aRow[4];
  #pragma unroll
  for (int i = 0; i < 4; ++i)
    aRow[i] = A + (size_t)(m_base + i * 16 + lr) * GK + koff;
  const _Float16* bRow[4];
  #pragma unroll
  for (int j = 0; j < 4; ++j)
    bRow[j] = B + (size_t)(n_base + j * 16 + lr) * GK + koff;

  v8f acc[4][4];
  #pragma unroll
  for (int i = 0; i < 4; ++i)
    #pragma unroll
    for (int j = 0; j < 4; ++j)
      acc[i][j] = (v8f){};

  for (int kk = 0; kk < GK; kk += 32) {
    Frag a[4], b[4];
    #pragma unroll
    for (int i = 0; i < 4; ++i) {
      a[i].u[0] = *(const v4u*)(aRow[i] + kk);        // K = kk+koff .. +7
      a[i].u[1] = *(const v4u*)(aRow[i] + kk + 16);   // K = kk+koff+16 .. +23
    }
    #pragma unroll
    for (int j = 0; j < 4; ++j) {
      b[j].u[0] = *(const v4u*)(bRow[j] + kk);
      b[j].u[1] = *(const v4u*)(bRow[j] + kk + 16);
    }
    #pragma unroll
    for (int i = 0; i < 4; ++i)
      #pragma unroll
      for (int j = 0; j < 4; ++j)
        acc[i][j] = __builtin_amdgcn_wmma_f32_16x16x32_f16(
            /*neg_a=*/false, a[i].h, /*neg_b=*/false, b[j].h,
            /*c_mod=*/(short)0, acc[i][j],
            /*reuse_a=*/false, /*reuse_b=*/false);
  }

  // C/D layout: VGPR r -> row r + 8*hi, col = lane&15
  #pragma unroll
  for (int i = 0; i < 4; ++i)
    #pragma unroll
    for (int j = 0; j < 4; ++j) {
      float* cp = C + (size_t)(m_base + i * 16 + hi * 8) * GN + n_base + j * 16 + lr;
      #pragma unroll
      for (int r = 0; r < 8; ++r)
        cp[(size_t)r * GN] = acc[i][j][r];
    }
}

// ------------------------------------------------------------------
extern "C" void kernel_launch(void* const* d_in, const int* in_sizes, int n_in,
                              void* d_out, int out_size, void* d_ws, size_t ws_size,
                              hipStream_t stream) {
  (void)in_sizes; (void)n_in; (void)out_size; (void)ws_size;

  const float* x = (const float*)d_in[0];   // [GM, GK] fp32
  const float* w = (const float*)d_in[1];   // [GN, GK] fp32
  float*     out = (float*)d_out;           // [GM, GN] fp32

  _Float16* xq = (_Float16*)d_ws;                    //  8.4 MB
  _Float16* wq = xq + (size_t)GM * GK;               // 33.6 MB

  const int xblk = GM * GK / 32;
  const int wblk = GN * GK / 32;
  nvfp4_quantdq_kernel<<<(xblk + 255) / 256, 256, 0, stream>>>(x, xq, xblk);
  nvfp4_quantdq_kernel<<<(wblk + 255) / 256, 256, 0, stream>>>(w, wq, wblk);

  dim3 grid(GM / 128, GN / 128);
  nvfp4_gemm_kernel<<<grid, 128, 0, stream>>>(xq, wq, out);
}
